// Manhattan_LSTM_40097814675553
// MI455X (gfx1250) — matrix operand order repaired and checked by reading, not compiled, not run on hardware
//
#include <hip/hip_runtime.h>
#include <math.h>

// ---------------------------------------------------------------------------
// Manhattan-LSTM for MI455X (gfx1250, wave32, WMMA).
//
// Pipeline:
//   K0 prep_pack   : swizzle Wih/Whh -> bf16 B-fragment layout, bias sums
//   K1 xg_gemm     : xg[s,d] = emb[tok_s] x Wih_d^T + b   (WMMA bf16)
//   K2 lstm_rec    : 4 persistent blocks (sent x dir), 64 sequential steps,
//                    gates = xg + h x Whh^T via WMMA, h double-buffered in LDS
//   K3 postproc    : maxpool / attention softmax / commonWords / sim
// ---------------------------------------------------------------------------

typedef __attribute__((ext_vector_type(16))) __bf16 v16bf;
typedef __attribute__((ext_vector_type(8)))  __bf16 v8bf;
typedef __attribute__((ext_vector_type(8)))  float  v8f;

#define SEQ   64
#define BATCH 128
#define EMB   512
#define HID   256
#define G4    1024              // 4*HID
#define MROWS 8192              // SEQ*BATCH

// workspace byte offsets
#define OFF_XG    ((size_t)0)                       // 4 * 8192*1024 f32 = 128 MiB
#define OFF_O1    ((size_t)134217728)               // 64*128*512 f32 = 16 MiB
#define OFF_O2    ((size_t)150994944)               // 16 MiB
#define OFF_WIHP  ((size_t)167772160)               // 2 * 524288 bf16 = 2 MiB
#define OFF_WHHP  ((size_t)169869312)               // 2 * 262144 bf16 = 1 MiB
#define OFF_BSUM  ((size_t)170917888)               // 2 * 1024 f32

#define HSTRIDE   264                               // padded bf16 row stride for h in LDS

__device__ __forceinline__ float sigf(float x) { return 1.0f / (1.0f + expf(-x)); }

// ---------------------------------------------------------------------------
// K0: pack weights into WMMA B-fragment order (lane-major, 16 bf16/lane) and
// fold the two bias vectors.
//   B fragment (32x16 bf16): lane l holds column n = nt*16 + (l&15),
//   K values k = kc*32 + (l>>4)*8 + (i&7) + (i>>3)*16  (i = 0..15, contiguous in memory)
// ---------------------------------------------------------------------------
__global__ void prep_pack(const float* __restrict__ WihF, const float* __restrict__ WhhF,
                          const float* __restrict__ bihF, const float* __restrict__ bhhF,
                          const float* __restrict__ WihB, const float* __restrict__ WhhB,
                          const float* __restrict__ bihB, const float* __restrict__ bhhB,
                          __bf16* __restrict__ wihP, __bf16* __restrict__ whhP,
                          float* __restrict__ bsum) {
  const int total = 2 * 524288 + 2 * 262144 + 2 * 1024;
  for (int idx = blockIdx.x * blockDim.x + threadIdx.x; idx < total;
       idx += gridDim.x * blockDim.x) {
    if (idx < 2 * 524288) {                                     // Wih: (1024 x 512)
      int d = idx >> 19, rem = idx & 524287;
      const float* W = d ? WihB : WihF;
      int blk = rem >> 9, e = rem & 511, lane = e >> 4, i = e & 15;
      int nt = blk >> 4, kc = blk & 15;                         // 64 ntiles x 16 kchunks
      int n = nt * 16 + (lane & 15);
      int k = kc * 32 + (lane >> 4) * 8 + (i & 7) + (i >> 3) * 16;
      wihP[(size_t)d * 524288 + rem] = (__bf16)W[n * EMB + k];
    } else if (idx < 2 * 524288 + 2 * 262144) {                 // Whh: (1024 x 256)
      int j = idx - 2 * 524288;
      int d = j >> 18, rem = j & 262143;
      const float* W = d ? WhhB : WhhF;
      int blk = rem >> 9, e = rem & 511, lane = e >> 4, i = e & 15;
      int nt = blk >> 3, kc = blk & 7;                          // 64 ntiles x 8 kchunks
      int n = nt * 16 + (lane & 15);
      int k = kc * 32 + (lane >> 4) * 8 + (i & 7) + (i >> 3) * 16;
      whhP[(size_t)d * 262144 + rem] = (__bf16)W[n * HID + k];
    } else {                                                    // bias sums
      int j = idx - 2 * 524288 - 2 * 262144;
      int d = j >> 10, g = j & 1023;
      bsum[d * G4 + g] = d ? (bihB[g] + bhhB[g]) : (bihF[g] + bhhF[g]);
    }
  }
}

// ---------------------------------------------------------------------------
// K1: xg[sd] (8192 x 1024) = bf16(emb[tok]) x bf16(Wih^T) + bias, fp32 out.
// grid = (512 mtiles, 4 sd); block = 256 (8 waves); wave -> 8 ntiles.
// ---------------------------------------------------------------------------
__global__ void xg_gemm(const int* __restrict__ t1, const int* __restrict__ t2,
                        const float* __restrict__ emb, const __bf16* __restrict__ wihP,
                        const float* __restrict__ bsum, float* __restrict__ XG) {
  const int sd = blockIdx.y, s = sd >> 1, d = sd & 1;
  const int* tok = s ? t2 : t1;
  const int mtile = blockIdx.x;
  const int wave = threadIdx.x >> 5, lane = threadIdx.x & 31;

  const int mrow = mtile * 16 + (lane & 15);              // A row for this lane
  const int token = tok[mrow];
  const float* xrow = emb + (size_t)token * EMB;
  const __bf16* wp = wihP + (size_t)d * 524288;
  float* xg = XG + (size_t)sd * MROWS * G4;

  v8f acc[8] = {};
  for (int kc = 0; kc < 16; ++kc) {
    // A fragment: lane (l&15)=row, (l>>4) selects K sub-half; two runs of 8 floats
    const int k0 = kc * 32 + (lane >> 4) * 8;
    float4 a0 = *(const float4*)(xrow + k0);
    float4 a1 = *(const float4*)(xrow + k0 + 4);
    float4 a2 = *(const float4*)(xrow + k0 + 16);
    float4 a3 = *(const float4*)(xrow + k0 + 20);
    v16bf a;
    a[0]  = (__bf16)a0.x; a[1]  = (__bf16)a0.y; a[2]  = (__bf16)a0.z; a[3]  = (__bf16)a0.w;
    a[4]  = (__bf16)a1.x; a[5]  = (__bf16)a1.y; a[6]  = (__bf16)a1.z; a[7]  = (__bf16)a1.w;
    a[8]  = (__bf16)a2.x; a[9]  = (__bf16)a2.y; a[10] = (__bf16)a2.z; a[11] = (__bf16)a2.w;
    a[12] = (__bf16)a3.x; a[13] = (__bf16)a3.y; a[14] = (__bf16)a3.z; a[15] = (__bf16)a3.w;
#pragma unroll
    for (int j = 0; j < 8; ++j) {
      const int nt = wave * 8 + j;
      v16bf bb = *(const v16bf*)(wp + ((size_t)(nt * 16 + kc) * 512) + lane * 16);
      acc[j] = __builtin_amdgcn_wmma_f32_16x16x32_bf16(false, a, false, bb,
                                                       (short)0, acc[j], false, false);
    }
  }
  const float* bs = bsum + d * G4;
#pragma unroll
  for (int j = 0; j < 8; ++j) {
    const int nt = wave * 8 + j;
    const int n = nt * 16 + (lane & 15);
    const float bv = bs[n];
#pragma unroll
    for (int r = 0; r < 8; ++r) {
      const int m = mtile * 16 + r + 8 * (lane >> 4);    // C/D: vgpr r -> row
      xg[(size_t)m * G4 + n] = acc[j][r] + bv;
    }
  }
}

// ---------------------------------------------------------------------------
// K2: recurrent LSTM. grid = 4 (sent x dir), block = 1024 (32 waves).
// h state double-buffered in LDS (bf16, padded stride); c state in registers.
// Wave (w) owns mtile = w/4 and hhtiles (w%4)*4 .. +3; for each task it
// computes all four gate quadrants so the i,f,g,o update is in-register.
// ---------------------------------------------------------------------------
__global__ void lstm_rec(const float* __restrict__ XG, const __bf16* __restrict__ whhP,
                         float* __restrict__ O1, float* __restrict__ O2) {
  extern __shared__ __bf16 hsh[];                        // 2 * 128 * HSTRIDE bf16
  const int sd = blockIdx.x, s = sd >> 1, dir = sd & 1;
  const float* xg = XG + (size_t)sd * MROWS * G4;
  const __bf16* wp = whhP + (size_t)dir * 262144;
  float* o = s ? O2 : O1;

  const int tid = threadIdx.x, wave = tid >> 5, lane = tid & 31;
  const int mtile = wave >> 2;
  const int hhbase = (wave & 3) * 4;
  const int mA = mtile * 16 + (lane & 15);               // A-fragment row in LDS

  for (int i = tid; i < 2 * 128 * HSTRIDE; i += 1024) hsh[i] = (__bf16)0.0f;
  __syncthreads();

  float c[4][8];
#pragma unroll
  for (int t4 = 0; t4 < 4; ++t4)
#pragma unroll
    for (int r = 0; r < 8; ++r) c[t4][r] = 0.0f;

#pragma unroll 1
  for (int t = 0; t < SEQ; ++t) {
    const int l = dir ? (SEQ - 1 - t) : t;
    const __bf16* hread = hsh + (size_t)(t & 1) * (128 * HSTRIDE);
    __bf16* hwrite = hsh + (size_t)((t + 1) & 1) * (128 * HSTRIDE);

#pragma unroll 1
    for (int task = 0; task < 4; ++task) {
      const int hhtile = hhbase + task;
      v8f acc[4];
      // seed accumulators with the precomputed input contribution (+bias)
#pragma unroll
      for (int q = 0; q < 4; ++q) {
        const int n = q * HID + hhtile * 16 + (lane & 15);
#pragma unroll
        for (int r = 0; r < 8; ++r) {
          const int m = mtile * 16 + r + 8 * (lane >> 4);
          acc[q][r] = xg[((size_t)l * BATCH + m) * G4 + n];
        }
      }
      // gates += h x Whh^T
#pragma unroll
      for (int kc = 0; kc < 8; ++kc) {
        const int k0 = kc * 32 + (lane >> 4) * 8;
        v8bf lo = *(const v8bf*)(hread + mA * HSTRIDE + k0);
        v8bf hi = *(const v8bf*)(hread + mA * HSTRIDE + k0 + 16);
        v16bf a;
#pragma unroll
        for (int i = 0; i < 8; ++i) { a[i] = lo[i]; a[i + 8] = hi[i]; }
#pragma unroll
        for (int q = 0; q < 4; ++q) {
          const int nt = q * 16 + hhtile;
          v16bf bb = *(const v16bf*)(wp + ((size_t)(nt * 8 + kc) * 512) + lane * 16);
          acc[q] = __builtin_amdgcn_wmma_f32_16x16x32_bf16(false, a, false, bb,
                                                           (short)0, acc[q], false, false);
        }
      }
      // elementwise cell update (torch gate order i,f,g,o)
      const int hh = hhtile * 16 + (lane & 15);
#pragma unroll
      for (int r = 0; r < 8; ++r) {
        const int m = mtile * 16 + r + 8 * (lane >> 4);
        const float ii = acc[0][r], ff = acc[1][r], gg = acc[2][r], oo = acc[3][r];
        const float cn = sigf(ff) * c[task][r] + sigf(ii) * tanhf(gg);
        const float hn = sigf(oo) * tanhf(cn);
        c[task][r] = cn;
        hwrite[m * HSTRIDE + hh] = (__bf16)hn;           // next step's A operand
        o[((size_t)l * BATCH + m) * (2 * HID) + dir * HID + hh] = hn;
      }
    }
    __syncthreads();                                     // publish h for step t+1
  }
}

// ---------------------------------------------------------------------------
// K3: pooling / attention / commonWords / similarity. One block per batch row.
// ---------------------------------------------------------------------------
__global__ void postproc(const float* __restrict__ O1, const float* __restrict__ O2,
                         const int* __restrict__ t1, const int* __restrict__ t2,
                         float* __restrict__ out) {
  const int b = blockIdx.x, tid = threadIdx.x;
  const int D = 2 * HID;
  __shared__ float mp[512], att[SEQ], np[512], red[256];
  __shared__ int maskA[SEQ], posA[SEQ], hasFlag;

  // mp1 = max over time of o1
  for (int dd = tid; dd < D; dd += 256) {
    float mx = -3.402823466e38f;
    for (int l = 0; l < SEQ; ++l) mx = fmaxf(mx, O1[((size_t)l * BATCH + b) * D + dd]);
    mp[dd] = mx;
  }
  if (tid == 0) hasFlag = 0;
  __syncthreads();

  // att[l'] = sum_d mp1[d] * raw_o2[b*D*L + d*L + l']   (faithful raw .view)
  if (tid < SEQ) {
    const float* o2b = O2 + (size_t)b * D * SEQ;
    float sum = 0.0f;
    for (int dd = 0; dd < D; ++dd) sum += mp[dd] * o2b[dd * SEQ + tid];
    att[tid] = sum;
  }
  __syncthreads();
  if (tid == 0) {                                        // softmax over L=64
    float mx = att[0];
    for (int i = 1; i < SEQ; ++i) mx = fmaxf(mx, att[i]);
    float sum = 0.0f;
    for (int i = 0; i < SEQ; ++i) { float e = expf(att[i] - mx); att[i] = e; sum += e; }
    const float inv = 1.0f / sum;
    for (int i = 0; i < SEQ; ++i) att[i] *= inv;
  }
  __syncthreads();

  // new_pool[d] = sum_l att[l] * raw_o2[b*L*D + l*D + d]
  for (int dd = tid; dd < D; dd += 256) {
    const float* o2b = O2 + (size_t)b * D * SEQ;
    float sum = 0.0f;
    for (int l = 0; l < SEQ; ++l) sum += att[l] * o2b[l * D + dd];
    np[dd] = sum;
  }
  __syncthreads();

  // sim = exp(-L1(mp1, new_pool))
  float part = 0.0f;
  for (int dd = tid; dd < D; dd += 256) part += fabsf(mp[dd] - np[dd]);
  red[tid] = part;
  __syncthreads();
  for (int off = 128; off > 0; off >>= 1) {
    if (tid < off) red[tid] += red[tid + off];
    __syncthreads();
  }
  if (tid == 0) out[b] = expf(-red[0]);

  // commonWords: last matching index in sen1 for each sen2 position
  if (tid < SEQ) {
    const int tv = t2[tid * BATCH + b];
    int dbest = -1;
    for (int j = 0; j < SEQ; ++j)
      if (t1[j * BATCH + b] == tv) dbest = j;
    const int mk = (dbest > 1) && (tv > 0);
    maskA[tid] = mk;
    posA[tid] = dbest < 0 ? 0 : (dbest > SEQ - 1 ? SEQ - 1 : dbest);
    if (mk) atomicOr(&hasFlag, 1);
  }
  __syncthreads();
  const int has = hasFlag;

  float* e1 = out + BATCH;
  float* e2 = out + BATCH + BATCH * D;
  for (int dd = tid; dd < D; dd += 256) {
    float m1 = -3.402823466e38f, m2 = -3.402823466e38f;
    for (int i = 0; i < SEQ; ++i) {
      if (maskA[i]) {
        m1 = fmaxf(m1, O1[((size_t)posA[i] * BATCH + b) * D + dd]);
        m2 = fmaxf(m2, O2[((size_t)i * BATCH + b) * D + dd]);
      }
    }
    e1[(size_t)b * D + dd] = has ? m1 : 0.0f;
    e2[(size_t)b * D + dd] = has ? m2 : 0.0f;
  }
}

// ---------------------------------------------------------------------------
extern "C" void kernel_launch(void* const* d_in, const int* in_sizes, int n_in,
                              void* d_out, int out_size, void* d_ws, size_t ws_size,
                              hipStream_t stream) {
  (void)in_sizes; (void)n_in; (void)out_size; (void)ws_size;
  const int*   t1   = (const int*)d_in[0];
  const int*   t2   = (const int*)d_in[1];
  const float* emb  = (const float*)d_in[2];
  const float* WihF = (const float*)d_in[3];
  const float* WhhF = (const float*)d_in[4];
  const float* bihF = (const float*)d_in[5];
  const float* bhhF = (const float*)d_in[6];
  const float* WihB = (const float*)d_in[7];
  const float* WhhB = (const float*)d_in[8];
  const float* bihB = (const float*)d_in[9];
  const float* bhhB = (const float*)d_in[10];

  char* ws = (char*)d_ws;
  float*  XG   = (float*)(ws + OFF_XG);
  float*  O1   = (float*)(ws + OFF_O1);
  float*  O2   = (float*)(ws + OFF_O2);
  __bf16* wihP = (__bf16*)(ws + OFF_WIHP);
  __bf16* whhP = (__bf16*)(ws + OFF_WHHP);
  float*  bsum = (float*)(ws + OFF_BSUM);
  float*  out  = (float*)d_out;

  prep_pack<<<dim3(1536), dim3(256), 0, stream>>>(WihF, WhhF, bihF, bhhF,
                                                  WihB, WhhB, bihB, bhhB,
                                                  wihP, whhP, bsum);
  xg_gemm<<<dim3(512, 4), dim3(256), 0, stream>>>(t1, t2, emb, wihP, bsum, XG);
  lstm_rec<<<dim3(4), dim3(1024), 2 * 128 * HSTRIDE * sizeof(__bf16), stream>>>(
      XG, whhP, O1, O2);
  postproc<<<dim3(BATCH), dim3(256), 0, stream>>>(O1, O2, t1, t2, out);
}
